// SwinDiTStage_50972671869645
// MI455X (gfx1250) — compile-verified, matrix operand order checked by reading, not run on hardware
//
#include <hip/hip_runtime.h>

typedef __attribute__((ext_vector_type(16))) __bf16 v16bf;
typedef __attribute__((ext_vector_type(8)))  float  v8f;
typedef __attribute__((ext_vector_type(2)))  float  f32x2;
typedef __attribute__((ext_vector_type(2)))  __bf16 b16x2;

union Frag { v16bf v; unsigned int u[8]; };

// single v_cvt_pk_bf16_f32: pack two floats into one dword of bf16
__device__ __forceinline__ unsigned int pack2(float a, float b) {
    f32x2 f = {a, b};
    b16x2 h = __builtin_convertvector(f, b16x2);
    union { b16x2 h; unsigned int u; } t; t.h = h;
    return t.u;
}
__device__ __forceinline__ unsigned short bfs(float a) {
    union { __bf16 h; unsigned short s; } t;
    t.h = (__bf16)a;
    return t.s;
}
__device__ __forceinline__ float bflo(unsigned int u) {
    union { unsigned int u; float f; } x; x.u = u << 16; return x.f;
}
__device__ __forceinline__ float bfhi(unsigned int u) {
    union { unsigned int u; float f; } x; x.u = u & 0xffff0000u; return x.f;
}

// ---------------------------------------------------------------------------
// Weight transpose + fp32 -> bf16:  WT[m][n][k] = W_m[k][n],  m in {q,k,v,o}
// ---------------------------------------------------------------------------
__global__ __launch_bounds__(256) void convert_weights(
    const float* __restrict__ Wq, const float* __restrict__ Wk,
    const float* __restrict__ Wv, const float* __restrict__ Wo,
    unsigned short* __restrict__ WT)
{
    int gid = blockIdx.x * 256 + threadIdx.x;      // 4*256*256 total
    int m   = gid >> 16;
    int idx = gid & 65535;
    int n   = idx >> 8;
    int k   = idx & 255;
    const float* W = (m == 0) ? Wq : (m == 1) ? Wk : (m == 2) ? Wv : Wo;
    WT[gid] = bfs(W[k * 256 + n]);                 // WT layout: m*65536 + n*256 + k
}

// ---------------------------------------------------------------------------
// Continuous relative position bias:  bias[h][i][j] = 16*sigmoid(MLP(tbl)[rpi])
// ---------------------------------------------------------------------------
__global__ __launch_bounds__(256) void build_bias(
    const float* __restrict__ w1, const float* __restrict__ b1,
    const float* __restrict__ w2, float* __restrict__ bias)
{
    __shared__ float s_tbl[225 * 8];
    int tid = threadIdx.x;
    if (tid < 225) {
        int a = tid / 15, bb = tid % 15;
        float c0 = (float)(a - 7) * (8.0f / 7.0f);
        float c1 = (float)(bb - 7) * (8.0f / 7.0f);
        float s0 = (float)((c0 > 0.f) - (c0 < 0.f));
        float s1 = (float)((c1 > 0.f) - (c1 < 0.f));
        float t0 = s0 * __log2f(fabsf(c0) + 1.0f) * (1.0f / 3.0f);
        float t1 = s1 * __log2f(fabsf(c1) + 1.0f) * (1.0f / 3.0f);
        float o[8] = {0.f, 0.f, 0.f, 0.f, 0.f, 0.f, 0.f, 0.f};
        for (int j = 0; j < 512; ++j) {
            float hv = fmaxf(t0 * w1[j] + t1 * w1[512 + j] + b1[j], 0.0f);
            #pragma unroll
            for (int hd = 0; hd < 8; ++hd) o[hd] += hv * w2[j * 8 + hd];
        }
        #pragma unroll
        for (int hd = 0; hd < 8; ++hd) s_tbl[tid * 8 + hd] = o[hd];
    }
    __syncthreads();
    for (int e = tid; e < 8 * 64 * 64; e += 256) {
        int h = e >> 12, ij = e & 4095, i = ij >> 6, j = ij & 63;
        int dy = (i >> 3) - (j >> 3) + 7;
        int dx = (i & 7) - (j & 7) + 7;
        float v = s_tbl[(dy * 15 + dx) * 8 + h];
        bias[e] = 16.0f / (1.0f + __expf(-v));
    }
}

// ---------------------------------------------------------------------------
// Fused per-window SwinV2 attention.  1 block = 1 window, 8 waves = 8 heads.
// ---------------------------------------------------------------------------
__global__ __launch_bounds__(256) void swin_attn(
    const float* __restrict__ x, const unsigned short* __restrict__ WT,
    const float* __restrict__ bias16, const float* __restrict__ bq,
    const float* __restrict__ bv, const float* __restrict__ bo,
    const float* __restrict__ logit_scale, float* __restrict__ out)
{
    __shared__ __align__(16) unsigned short s_x [64 * 256];    // x bf16; later probs h0..3
    __shared__ __align__(16) unsigned short s_p2[4 * 64 * 64]; // probs h4..7
    __shared__ __align__(16) unsigned short s_q [64 * 256];    // q raw -> normalized*scale
    __shared__ __align__(16) unsigned short s_k [64 * 256];    // k raw -> normalized; later ctx
    __shared__ __align__(16) unsigned short s_vT[256 * 64];    // v transposed [c][tok]

    const int tid    = threadIdx.x;
    const int lane   = tid & 31;
    const int wv     = tid >> 5;
    const int laneLo = lane & 15;
    const int laneHi = lane >> 4;

    const int w  = blockIdx.x;          // 2048 windows
    const int b  = w >> 8;
    const int wh = (w >> 4) & 15;
    const int ww = w & 15;

    // ---- 1) window partition: x -> bf16 LDS -------------------------------
    {
        int n  = tid >> 2;              // token 0..63
        int c0 = (tid & 3) * 64;        // channel chunk
        int iy = wh * 8 + (n >> 3), ix = ww * 8 + (n & 7);
        const float4* src4 = (const float4*)(x + (((size_t)(b * 128 + iy) * 128 + ix) << 8) + c0);
        #pragma unroll
        for (int i = 0; i < 16; i += 2) {
            float4 t0 = src4[i], t1 = src4[i + 1];
            uint4 pk;
            pk.x = pack2(t0.x, t0.y); pk.y = pack2(t0.z, t0.w);
            pk.z = pack2(t1.x, t1.y); pk.w = pack2(t1.z, t1.w);
            *(uint4*)&s_x[n * 256 + c0 + i * 4] = pk;
        }
    }
    __syncthreads();

    // ---- 2) QKV projections via WMMA (192 tiles / 8 waves) ----------------
    // Q,K computed as (X W)^T = W^T X^T  -> contiguous row-major b128 stores.
    // V computed as  X W                 -> contiguous store into s_vT[c][tok].
    #pragma unroll 1
    for (int t = 0; t < 24; ++t) {
        int j    = wv * 24 + t;
        int mat  = j >> 6;              // 0=Q 1=K 2=V
        int tile = j & 63;
        v8f acc = {0.f, 0.f, 0.f, 0.f, 0.f, 0.f, 0.f, 0.f};
        if (mat < 2) {
            int tt = tile >> 4, ct = tile & 15;        // token tile, channel tile
            const unsigned short* Wm = WT + mat * 65536;
            #pragma unroll
            for (int kt = 0; kt < 8; ++kt) {
                int kb = kt * 32;
                Frag a, bx;
                #pragma unroll
                for (int p = 0; p < 8; ++p) {
                    int ko = ((p & 4) << 2) + (laneHi << 3) + ((p & 3) << 1);
                    a.u[p]  = *(const unsigned int*)&Wm[(ct * 16 + laneLo) * 256 + kb + ko];
                    bx.u[p] = *(const unsigned int*)&s_x[(tt * 16 + laneLo) * 256 + kb + ko];
                }
                acc = __builtin_amdgcn_wmma_f32_16x16x32_bf16(false, a.v, false, bx.v,
                                                              (short)0, acc, false, false);
            }
            unsigned short* dst = (mat == 0) ? s_q : s_k;
            uint4 pk;
            pk.x = pack2(acc[0], acc[1]); pk.y = pack2(acc[2], acc[3]);
            pk.z = pack2(acc[4], acc[5]); pk.w = pack2(acc[6], acc[7]);
            *(uint4*)&dst[(tt * 16 + laneLo) * 256 + ct * 16 + (laneHi << 3)] = pk;
        } else {
            int mt = tile >> 4, nt = tile & 15;        // token rows, channel cols
            int col = nt * 16 + laneLo;
            #pragma unroll
            for (int kt = 0; kt < 8; ++kt) {
                int kb = kt * 32;
                Frag a, bw;
                #pragma unroll
                for (int p = 0; p < 8; ++p) {
                    int ko = ((p & 4) << 2) + (laneHi << 3) + ((p & 3) << 1);
                    a.u[p]  = *(const unsigned int*)&s_x[(mt * 16 + laneLo) * 256 + kb + ko];
                    bw.u[p] = *(const unsigned int*)&WT[2 * 65536 + col * 256 + kb + ko];
                }
                acc = __builtin_amdgcn_wmma_f32_16x16x32_bf16(false, a.v, false, bw.v,
                                                              (short)0, acc, false, false);
            }
            float bb = bv[col];
            uint4 pk;
            pk.x = pack2(acc[0] + bb, acc[1] + bb); pk.y = pack2(acc[2] + bb, acc[3] + bb);
            pk.z = pack2(acc[4] + bb, acc[5] + bb); pk.w = pack2(acc[6] + bb, acc[7] + bb);
            *(uint4*)&s_vT[col * 64 + mt * 16 + (laneHi << 3)] = pk;
        }
    }
    __syncthreads();

    // ---- 3) add bq, cosine-normalize q,k (logit scale folded into q) ------
    #pragma unroll 1
    for (int jj = 0; jj < 4; ++jj) {
        int job = tid + jj * 256;       // 1024 = 64 tok * 8 heads * {q,k}
        int isK = job >> 9;
        int tk  = job & 511;
        int tok = tk >> 3, hh = tk & 7;
        unsigned short* buf = isK ? s_k : s_q;
        int base = tok * 256 + hh * 32;
        uint4 raw[4];
        #pragma unroll
        for (int i = 0; i < 4; ++i) raw[i] = *(const uint4*)&buf[base + i * 8];
        float vals[32];
        #pragma unroll
        for (int i = 0; i < 4; ++i) {
            vals[i*8+0] = bflo(raw[i].x); vals[i*8+1] = bfhi(raw[i].x);
            vals[i*8+2] = bflo(raw[i].y); vals[i*8+3] = bfhi(raw[i].y);
            vals[i*8+4] = bflo(raw[i].z); vals[i*8+5] = bfhi(raw[i].z);
            vals[i*8+6] = bflo(raw[i].w); vals[i*8+7] = bfhi(raw[i].w);
        }
        if (!isK) {
            #pragma unroll
            for (int c = 0; c < 32; ++c) vals[c] += bq[hh * 32 + c];
        }
        float ss = 0.f;
        #pragma unroll
        for (int c = 0; c < 32; ++c) ss += vals[c] * vals[c];
        float inv = 1.0f / fmaxf(sqrtf(ss), 1e-12f);
        if (!isK) inv *= __expf(fminf(logit_scale[hh], 4.60517019f));   // ln(100)
        #pragma unroll
        for (int i = 0; i < 4; ++i) {
            uint4 pk;
            pk.x = pack2(vals[i*8+0] * inv, vals[i*8+1] * inv);
            pk.y = pack2(vals[i*8+2] * inv, vals[i*8+3] * inv);
            pk.z = pack2(vals[i*8+4] * inv, vals[i*8+5] * inv);
            pk.w = pack2(vals[i*8+6] * inv, vals[i*8+7] * inv);
            *(uint4*)&buf[base + i * 8] = pk;
        }
    }
    __syncthreads();

    // ---- 4) scores + bias + softmax (wave == head) ------------------------
    const int h = wv;
    v8f sc[16];
    #pragma unroll
    for (int mt = 0; mt < 4; ++mt) {
        #pragma unroll
        for (int ntk = 0; ntk < 4; ++ntk) {
            Frag a, bk;
            #pragma unroll
            for (int p = 0; p < 8; ++p) {
                int ko = ((p & 4) << 2) + (laneHi << 3) + ((p & 3) << 1);
                a.u[p]  = *(const unsigned int*)&s_q[(mt * 16 + laneLo) * 256 + h * 32 + ko];
                bk.u[p] = *(const unsigned int*)&s_k[(ntk * 16 + laneLo) * 256 + h * 32 + ko];
            }
            v8f acc = {0.f, 0.f, 0.f, 0.f, 0.f, 0.f, 0.f, 0.f};
            acc = __builtin_amdgcn_wmma_f32_16x16x32_bf16(false, a.v, false, bk.v,
                                                          (short)0, acc, false, false);
            const float* bp = bias16 + h * 4096 + (ntk * 16 + laneLo);
            #pragma unroll
            for (int r = 0; r < 8; ++r)
                acc[r] += bp[(mt * 16 + (laneHi << 3) + r) * 64];
            sc[mt * 4 + ntk] = acc;
        }
    }
    unsigned short* pb = ((h < 4) ? s_x : s_p2) + (h & 3) * 4096;
    #pragma unroll
    for (int mt = 0; mt < 4; ++mt) {
        float rmax[8], rinv[8];
        #pragma unroll
        for (int r = 0; r < 8; ++r) {
            float m = fmaxf(fmaxf(sc[mt * 4 + 0][r], sc[mt * 4 + 1][r]),
                            fmaxf(sc[mt * 4 + 2][r], sc[mt * 4 + 3][r]));
            #pragma unroll
            for (int d = 1; d < 16; d <<= 1) m = fmaxf(m, __shfl_xor(m, d, 32));
            rmax[r] = m;
        }
        #pragma unroll
        for (int r = 0; r < 8; ++r) {
            float s = 0.f;
            #pragma unroll
            for (int ntk = 0; ntk < 4; ++ntk) {
                float e = __expf(sc[mt * 4 + ntk][r] - rmax[r]);
                sc[mt * 4 + ntk][r] = e; s += e;
            }
            #pragma unroll
            for (int d = 1; d < 16; d <<= 1) s += __shfl_xor(s, d, 32);
            rinv[r] = 1.0f / s;
        }
        #pragma unroll
        for (int ntk = 0; ntk < 4; ++ntk)
            #pragma unroll
            for (int r = 0; r < 8; ++r)
                pb[(mt * 16 + (laneHi << 3) + r) * 64 + ntk * 16 + laneLo]
                    = bfs(sc[mt * 4 + ntk][r] * rinv[r]);
    }
    __syncthreads();

    // ---- 5) ctx^T = V^T P^T  -> contiguous b128 stores into s_k -----------
    {
        #pragma unroll
        for (int dt = 0; dt < 2; ++dt) {           // d-tile (rows of ctx^T)
            #pragma unroll
            for (int tt = 0; tt < 4; ++tt) {       // token tile (cols of ctx^T)
                v8f acc = {0.f, 0.f, 0.f, 0.f, 0.f, 0.f, 0.f, 0.f};
                #pragma unroll
                for (int kt = 0; kt < 2; ++kt) {
                    int kb = kt * 32;
                    Frag a, bp2;
                    #pragma unroll
                    for (int p = 0; p < 8; ++p) {
                        int ko = ((p & 4) << 2) + (laneHi << 3) + ((p & 3) << 1);
                        a.u[p]   = *(const unsigned int*)&s_vT[(h * 32 + dt * 16 + laneLo) * 64 + kb + ko];
                        bp2.u[p] = *(const unsigned int*)&pb[(tt * 16 + laneLo) * 64 + kb + ko];
                    }
                    acc = __builtin_amdgcn_wmma_f32_16x16x32_bf16(false, a.v, false, bp2.v,
                                                                  (short)0, acc, false, false);
                }
                uint4 pk;
                pk.x = pack2(acc[0], acc[1]); pk.y = pack2(acc[2], acc[3]);
                pk.z = pack2(acc[4], acc[5]); pk.w = pack2(acc[6], acc[7]);
                *(uint4*)&s_k[(tt * 16 + laneLo) * 256 + h * 32 + dt * 16 + (laneHi << 3)] = pk;
            }
        }
    }
    __syncthreads();

    // ---- 6) out = ctx @ Wo + bo, window-reverse store ---------------------
    #pragma unroll 1
    for (int t = 0; t < 8; ++t) {
        int tile = wv * 8 + t;          // 64 tiles
        int mt = tile >> 4, nt = tile & 15;
        int col = nt * 16 + laneLo;
        v8f acc = {0.f, 0.f, 0.f, 0.f, 0.f, 0.f, 0.f, 0.f};
        #pragma unroll
        for (int kt = 0; kt < 8; ++kt) {
            int kb = kt * 32;
            Frag a, bw;
            #pragma unroll
            for (int p = 0; p < 8; ++p) {
                int ko = ((p & 4) << 2) + (laneHi << 3) + ((p & 3) << 1);
                a.u[p]  = *(const unsigned int*)&s_k[(mt * 16 + laneLo) * 256 + kb + ko];
                bw.u[p] = *(const unsigned int*)&WT[3 * 65536 + col * 256 + kb + ko];
            }
            acc = __builtin_amdgcn_wmma_f32_16x16x32_bf16(false, a.v, false, bw.v,
                                                          (short)0, acc, false, false);
        }
        float bb = bo[col];
        #pragma unroll
        for (int r = 0; r < 8; ++r) {
            int q  = mt * 16 + (laneHi << 3) + r;
            int iy = wh * 8 + (q >> 3), ix = ww * 8 + (q & 7);
            out[(((size_t)(b * 128 + iy) * 128 + ix) << 8) + col] = acc[r] + bb;
        }
    }
}

// ---------------------------------------------------------------------------
extern "C" void kernel_launch(void* const* d_in, const int* in_sizes, int n_in,
                              void* d_out, int out_size, void* d_ws, size_t ws_size,
                              hipStream_t stream) {
    const float* x   = (const float*)d_in[0];
    const float* Wq  = (const float*)d_in[1];
    const float* bq  = (const float*)d_in[2];
    const float* Wk  = (const float*)d_in[3];
    const float* Wv  = (const float*)d_in[4];
    const float* bv  = (const float*)d_in[5];
    const float* Wo  = (const float*)d_in[6];
    const float* bo  = (const float*)d_in[7];
    const float* cw1 = (const float*)d_in[8];
    const float* cb1 = (const float*)d_in[9];
    const float* cw2 = (const float*)d_in[10];
    const float* ls  = (const float*)d_in[11];
    float* out = (float*)d_out;

    unsigned short* WT = (unsigned short*)d_ws;                       // 512 KB
    float* bias = (float*)((char*)d_ws + 4 * 65536 * sizeof(unsigned short)); // 128 KB

    convert_weights<<<1024, 256, 0, stream>>>(Wq, Wk, Wv, Wo, WT);
    build_bias<<<1, 256, 0, stream>>>(cw1, cb1, cw2, bias);
    swin_attn<<<2048, 256, 0, stream>>>(x, WT, bias, bq, bv, bo, ls, out);
}
